// ObjectMatchingModule_22771916603814
// MI455X (gfx1250) — matrix operand order repaired and checked by reading, not compiled, not run on hardware
//
#include <hip/hip_runtime.h>
#include <hip/hip_bf16.h>

typedef __attribute__((ext_vector_type(16))) __bf16 v16bf;
typedef __attribute__((ext_vector_type(8)))  float  v8f;

#define SINK_ITERS 100
#define MDIM 2048
#define NDIM 2048
#define DDIM 512
#define LDK  2049      // K is (M+1) x (N+1), contiguous in d_out
#define CS_PAD 2304    // 9*256, padded colsum length
#define NCHUNK 16      // row chunks for column reduction
#define CHUNK  129     // ceil(2049/16)

// Split fp32 into bf16 hi + bf16 lo (RNE rounding), packed pairwise as u32.
__device__ __forceinline__ void split2_bf16(float x, float y,
                                            unsigned &hi, unsigned &lo) {
  unsigned ux = __float_as_uint(x), uy = __float_as_uint(y);
  unsigned hx = ux + 0x7FFFu + ((ux >> 16) & 1u);
  unsigned hy = uy + 0x7FFFu + ((uy >> 16) & 1u);
  float rx = x - __uint_as_float(hx & 0xFFFF0000u);
  float ry = y - __uint_as_float(hy & 0xFFFF0000u);
  hi = (hx >> 16) | (hy & 0xFFFF0000u);
  unsigned ulx = __float_as_uint(rx), uly = __float_as_uint(ry);
  unsigned lx = ulx + 0x7FFFu + ((ulx >> 16) & 1u);
  unsigned ly = uly + 0x7FFFu + ((uly >> 16) & 1u);
  lo = (lx >> 16) | (ly & 0xFFFF0000u);
}

union FragU { unsigned u[8]; v16bf v; };

// ---------------------------------------------------------------------------
// K[:2048,:2048] = exp( (q @ r^T) * 10 ), via split-bf16 WMMA (3 wmma / k32)
// Block tile 128x128, 8 waves, wave w owns rows [16w,16w+16) x 8 col tiles.
// ---------------------------------------------------------------------------
__global__ __launch_bounds__(256)
void gemm_exp_kernel(const float* __restrict__ q, const float* __restrict__ r,
                     float* __restrict__ Kmat) {
  __shared__ unsigned aTh[128 * 17];  // A hi, bf16 pairs, stride 17 u32 (pad)
  __shared__ unsigned aTl[128 * 17];  // A lo
  __shared__ unsigned bTh[128 * 17];  // B^T(=r) hi
  __shared__ unsigned bTl[128 * 17];  // B^T(=r) lo

  const int tid  = threadIdx.x;
  const int w    = tid >> 5;          // wave id 0..7 (wave32)
  const int lane = tid & 31;
  const int half = lane >> 4;
  const int l15  = lane & 15;
  const int bm   = blockIdx.y * 128;
  const int bn   = blockIdx.x * 128;

  v8f acc[8];
  const v8f zero8 = {0.f,0.f,0.f,0.f,0.f,0.f,0.f,0.f};
#pragma unroll
  for (int t = 0; t < 8; ++t) acc[t] = zero8;

  for (int k0 = 0; k0 < DDIM; k0 += 32) {
    // stage 128x32 tiles of q and r as bf16 hi/lo pairs
#pragma unroll
    for (int e = 0; e < 8; ++e) {
      int pid = e * 256 + tid;        // 0..2047
      int row = pid >> 4;             // 0..127
      int pc  = pid & 15;             // pair column: k = k0 + 2*pc
      float2 qa = *(const float2*)(q + (size_t)(bm + row) * DDIM + k0 + 2 * pc);
      unsigned h, l;
      split2_bf16(qa.x, qa.y, h, l);
      aTh[row * 17 + pc] = h; aTl[row * 17 + pc] = l;
      float2 rb = *(const float2*)(r + (size_t)(bn + row) * DDIM + k0 + 2 * pc);
      split2_bf16(rb.x, rb.y, h, l);
      bTh[row * 17 + pc] = h; bTl[row * 17 + pc] = l;
    }
    __syncthreads();

    // A fragment (16x32 bf16): lane l15 = M, VGPR i pair index per ISA layout
    FragU ah, al;
    const int arow = (16 * w + l15) * 17;
#pragma unroll
    for (int i = 0; i < 8; ++i) {
      int pa = ((i >= 4) ? 8 : 0) + (i & 3) + 4 * half;  // K = 2*pa
      ah.u[i] = aTh[arow + pa];
      al.u[i] = aTl[arow + pa];
    }
#pragma unroll
    for (int tn = 0; tn < 8; ++tn) {
      // B fragment (32x16 bf16): N = l15, VGPR v holds K pair ((v>=4)?8:0) + (v&3) + 4*half
      FragU bh, bl;
      const int brow = (16 * tn + l15) * 17 + 4 * half;
#pragma unroll
      for (int v2 = 0; v2 < 8; ++v2) { bh.u[v2] = bTh[brow + ((v2 >= 4) ? 8 : 0) + (v2 & 3)]; bl.u[v2] = bTl[brow + ((v2 >= 4) ? 8 : 0) + (v2 & 3)]; }
      acc[tn] = __builtin_amdgcn_wmma_f32_16x16x32_bf16(false, ah.v, false, bh.v,
                                                        (short)0, acc[tn], false, false);
      acc[tn] = __builtin_amdgcn_wmma_f32_16x16x32_bf16(false, ah.v, false, bl.v,
                                                        (short)0, acc[tn], false, false);
      acc[tn] = __builtin_amdgcn_wmma_f32_16x16x32_bf16(false, al.v, false, bh.v,
                                                        (short)0, acc[tn], false, false);
    }
    __syncthreads();
  }

  // C/D layout: VGPR rr -> M = rr + 8*half (+16w), N = l15 (+16tn)
#pragma unroll
  for (int tn = 0; tn < 8; ++tn) {
#pragma unroll
    for (int rr = 0; rr < 8; ++rr) {
      int Mrow = bm + 16 * w + rr + 8 * half;
      int Ncol = bn + 16 * tn + l15;
      Kmat[(size_t)Mrow * LDK + Ncol] = expf(acc[tn][rr] * 10.0f);
    }
  }
}

// Border row/col of K = exp(z*10); colsum-inverse initialized to 1.
__global__ void init_border(float* __restrict__ Kmat, float* __restrict__ csinv,
                            const float* __restrict__ z) {
  int idx = blockIdx.x * 256 + threadIdx.x;
  float border = expf(z[0] * 10.0f);
  if (idx <= NDIM) {
    Kmat[(size_t)idx * LDK + NDIM]  = border;   // last column
    Kmat[(size_t)MDIM * LDK + idx] = border;    // last row
  }
  if (idx < CS_PAD) csinv[idx] = 1.0f;
}

// Fused: K[i][j] = (K[i][j]*csinv[j]) / rowsum_i. Row cached in registers.
__global__ __launch_bounds__(256)
void fused_row(float* __restrict__ Kmat, const float* __restrict__ csinv) {
  __shared__ float red[256];
  const int tid = threadIdx.x;
  float* Krow = Kmat + (size_t)blockIdx.x * LDK;
  float vals[9];
  float s = 0.f;
#pragma unroll
  for (int e = 0; e < 9; ++e) {
    int j = tid + 256 * e;
    float v = 0.f;
    if (j < LDK) v = Krow[j] * csinv[j];
    vals[e] = v;
    s += v;
  }
  red[tid] = s; __syncthreads();
  for (int st = 128; st > 0; st >>= 1) {
    if (tid < st) red[tid] += red[tid + st];
    __syncthreads();
  }
  const float inv = 1.0f / red[0];
#pragma unroll
  for (int e = 0; e < 9; ++e) {
    int j = tid + 256 * e;
    if (j < LDK) Krow[j] = vals[e] * inv;
  }
}

// Partial column sums: chunk of rows per blockIdx.y, lane-coalesced columns.
__global__ __launch_bounds__(256)
void col_partial(const float* __restrict__ Kmat, float* __restrict__ part) {
  const int j = blockIdx.x * 256 + threadIdx.x;   // < CS_PAD
  float s = 0.f;
  if (j < LDK) {
    int r0 = blockIdx.y * CHUNK;
    int r1 = r0 + CHUNK; if (r1 > LDK) r1 = LDK;
    for (int i = r0; i < r1; ++i) s += Kmat[(size_t)i * LDK + j];
  }
  part[(size_t)blockIdx.y * CS_PAD + j] = s;
}

// Deterministic combine: csinv[j] = 1 / sum_chunks part[c][j]
__global__ void col_combine(const float* __restrict__ part, float* __restrict__ csinv) {
  const int j = blockIdx.x * 256 + threadIdx.x;   // < CS_PAD
  float s = 0.f;
#pragma unroll
  for (int c = 0; c < NCHUNK; ++c) s += part[(size_t)c * CS_PAD + j];
  csinv[j] = 1.0f / s;                            // unused lanes: inf, never read
}

// Final column normalization; emit K in place and P = K[:2048,:2048].
__global__ void final_div(float* __restrict__ Kmat, const float* __restrict__ csinv,
                          float* __restrict__ P) {
  const int j = blockIdx.x * 256 + threadIdx.x;
  const int i = blockIdx.y;
  if (j >= LDK) return;
  float v = Kmat[(size_t)i * LDK + j] * csinv[j];
  Kmat[(size_t)i * LDK + j] = v;
  if (i < MDIM && j < NDIM) P[(size_t)i * NDIM + j] = v;
}

extern "C" void kernel_launch(void* const* d_in, const int* in_sizes, int n_in,
                              void* d_out, int out_size, void* d_ws, size_t ws_size,
                              hipStream_t stream) {
  (void)in_sizes; (void)n_in; (void)out_size; (void)ws_size;
  const float* q = (const float*)d_in[0];
  const float* r = (const float*)d_in[1];
  const float* z = (const float*)d_in[2];
  float* P    = (float*)d_out;
  float* Kmat = P + (size_t)MDIM * NDIM;          // K region of d_out as workspace
  float* csinv = (float*)d_ws;                    // CS_PAD floats
  float* part  = csinv + CS_PAD;                  // NCHUNK * CS_PAD floats

  gemm_exp_kernel<<<dim3(NDIM / 128, MDIM / 128), 256, 0, stream>>>(q, r, Kmat);
  init_border<<<9, 256, 0, stream>>>(Kmat, csinv, z);

  for (int t = 0; t < SINK_ITERS; ++t) {
    fused_row  <<<LDK, 256, 0, stream>>>(Kmat, csinv);
    col_partial<<<dim3(9, NCHUNK), 256, 0, stream>>>(Kmat, part);
    col_combine<<<9, 256, 0, stream>>>(part, csinv);
  }
  final_div<<<dim3(9, LDK), 256, 0, stream>>>(Kmat, csinv, P);
}
